// AtomAttentionEncoder_2199023255730
// MI455X (gfx1250) — matrix operand order repaired and checked by reading, not compiled, not run on hardware
//
#include <hip/hip_runtime.h>
#include <math.h>

typedef __attribute__((ext_vector_type(16))) _Float16 v16h;
typedef __attribute__((ext_vector_type(8)))  float    v8f;

#define N_ATOM 16384
#define N_TOKN 512
#define CATOM  128
#define CPAIR  16
#define CTOK   384
#define CS     384
#define CZ     128
#define CREF   132
#define NQW    32
#define NKW    128
#define NHEAD  4
#define DHEAD  32

__device__ __forceinline__ float sigf(float x) { return 1.f / (1.f + __expf(-x)); }

// ---------------- WMMA fragment loaders (contiguous k order) ----------------
// A 16x32 f16: lane L -> row = m0 + (L&15); hi = L>>4.
// VGPR j: K = hi*16 + 2j + {0,1}
template <bool GUARD, bool RELU>
__device__ __forceinline__ v16h load_a(const float* __restrict__ A, int lda,
                                       int m0, int k0, int K, int lane) {
  const int r  = m0 + (lane & 15);
  const int hi = lane >> 4;
  const float* Ar = A + (size_t)r * lda + k0;
  v16h a;
#pragma unroll
  for (int j = 0; j < 8; ++j) {
    int kb = hi * 16 + 2 * j;
    float x0, x1;
    if (GUARD) {
      x0 = (k0 + kb     < K) ? Ar[kb]     : 0.f;
      x1 = (k0 + kb + 1 < K) ? Ar[kb + 1] : 0.f;
    } else {
      float2 xy = *(const float2*)(Ar + kb);   // kb even, rows 8B aligned
      x0 = xy.x; x1 = xy.y;
    }
    if (RELU) { x0 = fmaxf(x0, 0.f); x1 = fmaxf(x1, 0.f); }
    a[2 * j]     = (_Float16)x0;
    a[2 * j + 1] = (_Float16)x1;
  }
  return a;
}

// B 32x16 f16 (K x N, row-major source): lane L -> col n = n0 + (L&15); hi = L>>4.
// VGPR j: K = hi*16 + 2j + {0,1}
template <bool GUARD>
__device__ __forceinline__ v16h load_b(const float* __restrict__ B, int ldb,
                                       int k0, int n0, int K, int lane) {
  const int n  = n0 + (lane & 15);
  const int hi = lane >> 4;
  v16h b;
#pragma unroll
  for (int j = 0; j < 8; ++j) {
    int k = k0 + hi * 16 + 2 * j;
    float x0, x1;
    if (GUARD) {
      x0 = (k     < K) ? B[(size_t)k * ldb + n]       : 0.f;
      x1 = (k + 1 < K) ? B[(size_t)(k + 1) * ldb + n] : 0.f;
    } else {
      x0 = B[(size_t)k * ldb + n];
      x1 = B[(size_t)(k + 1) * ldb + n];
    }
    b[2 * j]     = (_Float16)x0;
    b[2 * j + 1] = (_Float16)x1;
  }
  return b;
}

// ---------------- Generic GEMM: C[M,N] = act(A[M,K] @ B[K,N]) ----------------
// Register-blocked MTx NT fragments per wave. ACTIN: 1 = relu(A). ACTOUT: 0/1/2.
template <int MT, int NT, int ACTIN, int ACTOUT>
__global__ void gemm_t(const float* __restrict__ A, const float* __restrict__ B,
                       float* __restrict__ C, int M, int N, int K,
                       int lda, int ldb, int ldc) {
  const int lane = threadIdx.x & 31;
  const int wid  = blockIdx.x * (blockDim.x >> 5) + (threadIdx.x >> 5);
  const int ntb  = N / (16 * NT);
  const int tiles = (M / (16 * MT)) * ntb;
  if (wid >= tiles) return;
  const int m0 = (wid / ntb) * 16 * MT;
  const int n0 = (wid % ntb) * 16 * NT;
  v8f zero8 = {};
  v8f acc[MT][NT];
#pragma unroll
  for (int mi = 0; mi < MT; ++mi)
#pragma unroll
    for (int ni = 0; ni < NT; ++ni) acc[mi][ni] = zero8;

  const int kmain = K & ~31;
  int k0 = 0;
  for (; k0 < kmain; k0 += 32) {
    v16h af[MT]; v16h bf[NT];
#pragma unroll
    for (int mi = 0; mi < MT; ++mi)
      af[mi] = load_a<false, ACTIN == 1>(A, lda, m0 + 16 * mi, k0, K, lane);
#pragma unroll
    for (int ni = 0; ni < NT; ++ni)
      bf[ni] = load_b<false>(B, ldb, k0, n0 + 16 * ni, K, lane);
#pragma unroll
    for (int mi = 0; mi < MT; ++mi)
#pragma unroll
      for (int ni = 0; ni < NT; ++ni)
        acc[mi][ni] = __builtin_amdgcn_wmma_f32_16x16x32_f16(
            false, af[mi], false, bf[ni], (short)0, acc[mi][ni], false, false);
  }
  if (k0 < K) {                       // ragged tail (only the K=132 GEMM hits this)
    v16h af[MT]; v16h bf[NT];
#pragma unroll
    for (int mi = 0; mi < MT; ++mi)
      af[mi] = load_a<true, ACTIN == 1>(A, lda, m0 + 16 * mi, k0, K, lane);
#pragma unroll
    for (int ni = 0; ni < NT; ++ni)
      bf[ni] = load_b<true>(B, ldb, k0, n0 + 16 * ni, K, lane);
#pragma unroll
    for (int mi = 0; mi < MT; ++mi)
#pragma unroll
      for (int ni = 0; ni < NT; ++ni)
        acc[mi][ni] = __builtin_amdgcn_wmma_f32_16x16x32_f16(
            false, af[mi], false, bf[ni], (short)0, acc[mi][ni], false, false);
  }

  const int hi = lane >> 4;
  const int nl = lane & 15;
#pragma unroll
  for (int mi = 0; mi < MT; ++mi)
#pragma unroll
    for (int ni = 0; ni < NT; ++ni)
#pragma unroll
      for (int v = 0; v < 8; ++v) {
        int m = m0 + 16 * mi + v + hi * 8;
        int n = n0 + 16 * ni + nl;
        float x = acc[mi][ni][v];
        if (ACTOUT == 1) x = fmaxf(x, 0.f);
        else if (ACTOUT == 2) x = sigf(x);
        C[(size_t)m * ldc + n] = x;
      }
}

// ---------------- LayerNorm (affine), wave per row ----------------
__global__ void ln_affine_k(const float* __restrict__ x, const float* __restrict__ g,
                            const float* __restrict__ b, float* __restrict__ out,
                            int rows, int cols) {
  int row = blockIdx.x * (blockDim.x >> 5) + (threadIdx.x >> 5);
  int lane = threadIdx.x & 31;
  if (row >= rows) return;
  const float* xr = x + (size_t)row * cols;
  float s = 0.f;
  for (int c = lane; c < cols; c += 32) s += xr[c];
#pragma unroll
  for (int off = 16; off >= 1; off >>= 1) s += __shfl_xor(s, off, 32);
  float mean = s / (float)cols;
  float vs = 0.f;
  for (int c = lane; c < cols; c += 32) { float d = xr[c] - mean; vs += d * d; }
#pragma unroll
  for (int off = 16; off >= 1; off >>= 1) vs += __shfl_xor(vs, off, 32);
  float inv = rsqrtf(vs / (float)cols + 1e-5f);
  float* o = out + (size_t)row * cols;
  for (int c = lane; c < cols; c += 32) o[c] = (xr[c] - mean) * inv * g[c] + b[c];
}

// ---------------- AdaLN: out = ln_na(q) * ag + ab (cols = 128) ----------------
__global__ void adaln_k(const float* __restrict__ q, const float* __restrict__ ag,
                        const float* __restrict__ ab, float* __restrict__ out, int rows) {
  int row = blockIdx.x * (blockDim.x >> 5) + (threadIdx.x >> 5);
  int lane = threadIdx.x & 31;
  if (row >= rows) return;
  const float* qr = q + (size_t)row * CATOM;
  float x[4]; float s = 0.f;
#pragma unroll
  for (int i = 0; i < 4; ++i) { x[i] = qr[lane + 32 * i]; s += x[i]; }
#pragma unroll
  for (int off = 16; off >= 1; off >>= 1) s += __shfl_xor(s, off, 32);
  float mean = s * (1.f / 128.f);
  float vs = 0.f;
#pragma unroll
  for (int i = 0; i < 4; ++i) { float d = x[i] - mean; vs += d * d; }
#pragma unroll
  for (int off = 16; off >= 1; off >>= 1) vs += __shfl_xor(vs, off, 32);
  float inv = rsqrtf(vs * (1.f / 128.f) + 1e-5f);
  const float* agr = ag + (size_t)row * CATOM;
  const float* abr = ab + (size_t)row * CATOM;
  float* o = out + (size_t)row * CATOM;
#pragma unroll
  for (int i = 0; i < 4; ++i) {
    int c = lane + 32 * i;
    o[c] = (x[i] - mean) * inv * agr[c] + abr[c];
  }
}

// ---------------- fused LN(z)*g+b then @ w_z -> zp[T*T,16], wave per row ----------------
__global__ void zproj_k(const float* __restrict__ z, const float* __restrict__ gz,
                        const float* __restrict__ bz, const float* __restrict__ wz,
                        float* __restrict__ zp, int nrows) {
  __shared__ float wzs[CZ * CPAIR];
  for (int i = threadIdx.x; i < CZ * CPAIR; i += blockDim.x) wzs[i] = wz[i];
  __syncthreads();
  int row = blockIdx.x * (blockDim.x >> 5) + (threadIdx.x >> 5);
  int lane = threadIdx.x & 31;
  if (row >= nrows) return;
  const float* zr = z + (size_t)row * CZ;
  float x[4]; float s = 0.f;
#pragma unroll
  for (int i = 0; i < 4; ++i) { x[i] = zr[lane + 32 * i]; s += x[i]; }
#pragma unroll
  for (int off = 16; off >= 1; off >>= 1) s += __shfl_xor(s, off, 32);
  float mean = s * (1.f / 128.f);
  float vs = 0.f;
#pragma unroll
  for (int i = 0; i < 4; ++i) { float d = x[i] - mean; vs += d * d; }
#pragma unroll
  for (int off = 16; off >= 1; off >>= 1) vs += __shfl_xor(vs, off, 32);
  float inv = rsqrtf(vs * (1.f / 128.f) + 1e-5f);
  float pj[16];
#pragma unroll
  for (int j = 0; j < 16; ++j) pj[j] = 0.f;
#pragma unroll
  for (int i = 0; i < 4; ++i) {
    int k = lane + 32 * i;
    float zn = (x[i] - mean) * inv * gz[k] + bz[k];
#pragma unroll
    for (int j = 0; j < 16; ++j) pj[j] += zn * wzs[k * 16 + j];
  }
#pragma unroll
  for (int off = 16; off >= 1; off >>= 1)
#pragma unroll
    for (int j = 0; j < 16; ++j) pj[j] += __shfl_xor(pj[j], off, 32);
  if (lane == 0) {
#pragma unroll
    for (int j = 0; j < 16; ++j) zp[(size_t)row * 16 + j] = pj[j];
  }
}

// ---------------- c += sproj[tok[row]] ----------------
__global__ void add_sproj_k(float* __restrict__ c, const float* __restrict__ sproj,
                            const int* __restrict__ tok) {
  int idx = blockIdx.x * blockDim.x + threadIdx.x;
  int row = idx >> 7, col = idx & 127;
  c[idx] += sproj[(size_t)tok[row] * CATOM + col];
}

// ---------------- q = c + r_l @ w_r ----------------
__global__ void q_init_k(const float* __restrict__ c, const float* __restrict__ r_l,
                         const float* __restrict__ w_r, float* __restrict__ q) {
  int idx = blockIdx.x * blockDim.x + threadIdx.x;
  int row = idx >> 7, col = idx & 127;
  float r0 = r_l[row * 3], r1 = r_l[row * 3 + 1], r2 = r_l[row * 3 + 2];
  q[idx] = c[idx] + r0 * w_r[col] + r1 * w_r[128 + col] + r2 * w_r[256 + col];
}

// ---------------- fused pair features + pair MLP (thread per (b,i,j)) ----------------
__global__ void pair_k(const float* __restrict__ ref_pos, const int* __restrict__ tok,
                       const float* __restrict__ zp, const float* __restrict__ cqt,
                       const float* __restrict__ ckt,
                       const float* __restrict__ w_dist, const float* __restrict__ w_invd,
                       const float* __restrict__ w_mask,
                       const float* __restrict__ w_p1, const float* __restrict__ w_p2,
                       const float* __restrict__ w_p3, float* __restrict__ p) {
  __shared__ float swd[48], swi[16], swm[16], sp1[256], sp2[256], sp3[256];
  for (int i = threadIdx.x; i < 48; i += blockDim.x) swd[i] = w_dist[i];
  for (int i = threadIdx.x; i < 16; i += blockDim.x) { swi[i] = w_invd[i]; swm[i] = w_mask[i]; }
  for (int i = threadIdx.x; i < 256; i += blockDim.x) {
    sp1[i] = w_p1[i]; sp2[i] = w_p2[i]; sp3[i] = w_p3[i];
  }
  __syncthreads();
  int t = blockIdx.x * blockDim.x + threadIdx.x;     // [0, 512*32*128)
  int j  = t & 127;
  int qi = t >> 7;                                   // atom query index
  int b  = qi >> 5;
  int kidx = b * NQW - 48 + j;
  int valid = (kidx >= 0 && kidx < N_ATOM);
  int kcl = kidx < 0 ? 0 : (kidx > N_ATOM - 1 ? N_ATOM - 1 : kidx);
  float dx = ref_pos[qi * 3 + 0] - ref_pos[kcl * 3 + 0];
  float dy = ref_pos[qi * 3 + 1] - ref_pos[kcl * 3 + 1];
  float dz = ref_pos[qi * 3 + 2] - ref_pos[kcl * 3 + 2];
  float inv = 1.f / (1.f + dx * dx + dy * dy + dz * dz);
  int uq = tok[qi], uk = tok[kcl];
  float vf = (valid && uq == uk) ? 1.f : 0.f;
  const float* zr = zp + ((size_t)uq * N_TOKN + uk) * CPAIR;
  const float* cq = cqt + (size_t)qi * CPAIR;
  const float* ck = ckt + (size_t)kcl * CPAIR;
  float pv[16];
#pragma unroll
  for (int c = 0; c < 16; ++c) {
    float base = dx * swd[c] + dy * swd[16 + c] + dz * swd[32 + c] + inv * swi[c] + swm[c];
    pv[c] = base * vf + zr[c] + cq[c] + ck[c];
  }
  float u[16], v1[16], v2[16];
#pragma unroll
  for (int c = 0; c < 16; ++c) u[c] = fmaxf(pv[c], 0.f);
#pragma unroll
  for (int o = 0; o < 16; ++o) {
    float a = 0.f;
#pragma unroll
    for (int c = 0; c < 16; ++c) a += u[c] * sp1[c * 16 + o];
    v1[o] = fmaxf(a, 0.f);
  }
#pragma unroll
  for (int o = 0; o < 16; ++o) {
    float a = 0.f;
#pragma unroll
    for (int c = 0; c < 16; ++c) a += v1[c] * sp2[c * 16 + o];
    v2[o] = fmaxf(a, 0.f);
  }
  float* po = p + (size_t)t * CPAIR;
#pragma unroll
  for (int o = 0; o < 16; ++o) {
    float a = 0.f;
#pragma unroll
    for (int c = 0; c < 16; ++c) a += v2[c] * sp3[c * 16 + o];
    po[o] = pv[o] + a;
  }
}

// ---------------- local pair-biased attention, block per window, wave per head --------
__global__ void attn_k(const float* __restrict__ qh, const float* __restrict__ kh,
                       const float* __restrict__ vh, const float* __restrict__ gs,
                       const float* __restrict__ p, const float* __restrict__ wbp,
                       float* __restrict__ ot) {
  extern __shared__ float smem[];                    // [4][32][128] logits/weights
  const int lane = threadIdx.x & 31;
  const int h    = threadIdx.x >> 5;
  const int b    = blockIdx.x;
  const int kbase = b * NQW - 48;
  const int hi = lane >> 4;
  const int nl = lane & 15;
  const float scale = 0.17677669529663687f;          // 1/sqrt(32)
  float* llog = smem + h * NQW * NKW;

  // logits = Q @ K^T * scale  (per head)
  v16h a0 = load_a<false, false>(qh + h * DHEAD, CATOM, b * NQW,      0, DHEAD, lane);
  v16h a1 = load_a<false, false>(qh + h * DHEAD, CATOM, b * NQW + 16, 0, DHEAD, lane);
#pragma unroll
  for (int kt = 0; kt < 8; ++kt) {
    int jcol = kt * 16 + nl;
    int krow = kbase + jcol;
    krow = krow < 0 ? 0 : (krow > N_ATOM - 1 ? N_ATOM - 1 : krow);
    const float* src = kh + (size_t)krow * CATOM + h * DHEAD;
    v16h bf;
#pragma unroll
    for (int j = 0; j < 8; ++j) {
      float2 xy = *(const float2*)(src + hi * 16 + 2 * j);
      bf[2 * j]     = (_Float16)xy.x;
      bf[2 * j + 1] = (_Float16)xy.y;
    }
    v8f c0 = {}; v8f c1 = {};
    c0 = __builtin_amdgcn_wmma_f32_16x16x32_f16(false, a0, false, bf, (short)0, c0, false, false);
    c1 = __builtin_amdgcn_wmma_f32_16x16x32_f16(false, a1, false, bf, (short)0, c1, false, false);
#pragma unroll
    for (int v = 0; v < 8; ++v) {
      llog[(v + hi * 8) * NKW + jcol]      = c0[v] * scale;
      llog[(16 + v + hi * 8) * NKW + jcol] = c1[v] * scale;
    }
  }
  __syncthreads();

  // bias from p @ wbp, mask, softmax  (thread per (head,row))
  {
    int i   = threadIdx.x & 31;
    int hh2 = threadIdx.x >> 5;
    float* row = smem + hh2 * NQW * NKW + i * NKW;
    float wb[16];
#pragma unroll
    for (int c = 0; c < 16; ++c) wb[c] = wbp[c * NHEAD + hh2];
    const float* prow = p + ((size_t)(b * NQW + i) * NKW) * CPAIR;
    float mx = -3.0e38f;
    for (int j = 0; j < NKW; ++j) {
      int kidx = kbase + j;
      float val;
      if (kidx >= 0 && kidx < N_ATOM) {
        const float* pj = prow + j * CPAIR;
        float bias = 0.f;
#pragma unroll
        for (int c = 0; c < 16; ++c) bias += pj[c] * wb[c];
        val = row[j] + bias;
      } else {
        val = -1.0e9f;
      }
      row[j] = val;
      mx = fmaxf(mx, val);
    }
    float ssum = 0.f;
    for (int j = 0; j < NKW; ++j) { float e = __expf(row[j] - mx); row[j] = e; ssum += e; }
    float rinv = 1.f / ssum;
    for (int j = 0; j < NKW; ++j) row[j] *= rinv;
  }
  __syncthreads();

  // o = w @ V, gated store
  v8f zero8 = {};
  v8f acc[2][2];
#pragma unroll
  for (int qt = 0; qt < 2; ++qt)
#pragma unroll
    for (int dt = 0; dt < 2; ++dt) acc[qt][dt] = zero8;
#pragma unroll
  for (int kc = 0; kc < 4; ++kc) {
    v16h aw[2];
#pragma unroll
    for (int qt = 0; qt < 2; ++qt) {
      const float* src = llog + (qt * 16 + nl) * NKW + kc * 32;
      v16h t;
#pragma unroll
      for (int j = 0; j < 8; ++j) {
        int kb = hi * 16 + 2 * j;
        float2 xy = *(const float2*)(src + kb);
        t[2 * j]     = (_Float16)xy.x;
        t[2 * j + 1] = (_Float16)xy.y;
      }
      aw[qt] = t;
    }
#pragma unroll
    for (int dt = 0; dt < 2; ++dt) {
      int dcol = h * DHEAD + dt * 16 + nl;
      v16h bf;
#pragma unroll
      for (int j = 0; j < 8; ++j) {
        int kk = hi * 16 + 2 * j;
        int r0 = kbase + kc * 32 + kk;
        int r0c = r0 < 0 ? 0 : (r0 > N_ATOM - 1 ? N_ATOM - 1 : r0);
        int r1c = (r0 + 1) < 0 ? 0 : ((r0 + 1) > N_ATOM - 1 ? N_ATOM - 1 : (r0 + 1));
        bf[2 * j]     = (_Float16)vh[(size_t)r0c * CATOM + dcol];
        bf[2 * j + 1] = (_Float16)vh[(size_t)r1c * CATOM + dcol];
      }
#pragma unroll
      for (int qt = 0; qt < 2; ++qt)
        acc[qt][dt] = __builtin_amdgcn_wmma_f32_16x16x32_f16(false, aw[qt], false, bf,
                                                             (short)0, acc[qt][dt],
                                                             false, false);
    }
  }
#pragma unroll
  for (int qt = 0; qt < 2; ++qt)
#pragma unroll
    for (int dt = 0; dt < 2; ++dt)
#pragma unroll
      for (int v = 0; v < 8; ++v) {
        int m = qt * 16 + v + hi * 8;
        int arow = b * NQW + m;
        int col = h * DHEAD + dt * 16 + nl;
        float g = gs[(size_t)arow * CATOM + col];
        ot[(size_t)arow * CATOM + col] = acc[qt][dt][v] * g;
      }
}

// ---------------- residual with gate, SwiGLU ----------------
__global__ void axpy_gate_k(float* __restrict__ q, const float* __restrict__ t,
                            const float* __restrict__ g, int n) {
  int idx = blockIdx.x * blockDim.x + threadIdx.x;
  if (idx < n) q[idx] += t[idx] * g[idx];
}
__global__ void silu_mul_k(float* __restrict__ h1, const float* __restrict__ h2, int n) {
  int idx = blockIdx.x * blockDim.x + threadIdx.x;
  if (idx < n) { float x = h1[idx]; h1[idx] = x * sigf(x) * h2[idx]; }
}

// ---------------- deterministic segment mean via binary search ----------------
__device__ __forceinline__ int lb_dev(const int* a, int n, int v) {
  int lo = 0, hi = n;
  while (lo < hi) { int m = (lo + hi) >> 1; if (a[m] < v) lo = m + 1; else hi = m; }
  return lo;
}
__global__ void segmean_k(const float* __restrict__ t, const int* __restrict__ tok,
                          float* __restrict__ atok) {
  int tk = blockIdx.x;
  int lo = lb_dev(tok, N_ATOM, tk);
  int hi = lb_dev(tok, N_ATOM, tk + 1);
  float cnt = fmaxf((float)(hi - lo), 1.f);
  for (int c = threadIdx.x; c < CTOK; c += blockDim.x) {
    float s = 0.f;
    for (int r = lo; r < hi; ++r) s += t[(size_t)r * CTOK + c];
    atok[(size_t)tk * CTOK + c] = s / cnt;
  }
}

extern "C" void kernel_launch(void* const* d_in, const int* in_sizes, int n_in,
                              void* d_out, int out_size, void* d_ws, size_t ws_size,
                              hipStream_t stream) {
  (void)in_sizes; (void)n_in; (void)out_size; (void)ws_size;
  const float* ref_pos   = (const float*)d_in[0];
  const float* ref_feats = (const float*)d_in[1];
  const float* r_l       = (const float*)d_in[2];
  const float* s_in      = (const float*)d_in[3];
  const float* z_in      = (const float*)d_in[4];
  const int*   tok_idx   = (const int*)  d_in[5];
  const float* w_feat = (const float*)d_in[6];
  const float* w_dist = (const float*)d_in[7];
  const float* w_invd = (const float*)d_in[8];
  const float* w_mask = (const float*)d_in[9];
  const float* g_s = (const float*)d_in[10];
  const float* b_s = (const float*)d_in[11];
  const float* w_s = (const float*)d_in[12];
  const float* g_z = (const float*)d_in[13];
  const float* b_z = (const float*)d_in[14];
  const float* w_z = (const float*)d_in[15];
  const float* w_r = (const float*)d_in[16];
  const float* w_cq = (const float*)d_in[17];
  const float* w_ck = (const float*)d_in[18];
  const float* w_p1 = (const float*)d_in[19];
  const float* w_p2 = (const float*)d_in[20];
  const float* w_p3 = (const float*)d_in[21];
  const float* wq  = (const float*)d_in[22];
  const float* wk  = (const float*)d_in[23];
  const float* wv  = (const float*)d_in[24];
  const float* wg  = (const float*)d_in[25];
  const float* wbp = (const float*)d_in[26];
  const float* wo  = (const float*)d_in[27];
  const float* ada_att_g = (const float*)d_in[28];
  const float* ada_att_b = (const float*)d_in[29];
  const float* gate_att  = (const float*)d_in[30];
  const float* w1 = (const float*)d_in[31];
  const float* w2 = (const float*)d_in[32];
  const float* w3 = (const float*)d_in[33];
  const float* ada_tr_g = (const float*)d_in[34];
  const float* ada_tr_b = (const float*)d_in[35];
  const float* gate_tr  = (const float*)d_in[36];
  const float* w_qout   = (const float*)d_in[37];

  float* out  = (float*)d_out;
  float* atok = out;                                   // [512,384]
  float* q    = out + (size_t)N_TOKN * CTOK;           // [16384,128]
  float* c    = q + (size_t)N_ATOM * CATOM;            // [16384,128]
  float* p    = c + (size_t)N_ATOM * CATOM;            // [512,32,128,16]

  float* wsf = (float*)d_ws;
  size_t off = 0;
  auto alloc = [&](size_t n) { float* r = wsf + off; off += n; return r; };
  float* s_ln  = alloc((size_t)N_TOKN * CS);
  float* sproj = alloc((size_t)N_TOKN * CATOM);
  float* zp    = alloc((size_t)N_TOKN * N_TOKN * CPAIR);
  float* cqt   = alloc((size_t)N_ATOM * CPAIR);
  float* ckt   = alloc((size_t)N_ATOM * CPAIR);
  float* abuf  = alloc((size_t)N_ATOM * CATOM);
  float* qhb   = alloc((size_t)N_ATOM * CATOM);
  float* khb   = alloc((size_t)N_ATOM * CATOM);
  float* vhb   = alloc((size_t)N_ATOM * CATOM);
  float* gsb   = alloc((size_t)N_ATOM * CATOM);
  float* agb   = alloc((size_t)N_ATOM * CATOM);
  float* abb   = alloc((size_t)N_ATOM * CATOM);
  float* gateb = alloc((size_t)N_ATOM * CATOM);
  float* otb   = alloc((size_t)N_ATOM * CATOM);
  float* tmpb  = alloc((size_t)N_ATOM * CATOM);
  float* h1b   = alloc((size_t)N_ATOM * 2 * CATOM);
  float* h2b   = alloc((size_t)N_ATOM * 2 * CATOM);
  float* tbuf  = alloc((size_t)N_ATOM * CTOK);

  // dispatch: 32x32 register-blocked path for the big GEMMs, 16x16 + input-ReLU
  // path for the N=16 projections (cqt/ckt).
  auto gemm = [&](const float* A, const float* B, float* C_, int M, int N, int K,
                  int lda, int ldb, int ldc, int actin, int actout) {
    if (actin == 0 && (M % 32 == 0) && (N % 32 == 0)) {
      int tiles = (M >> 5) * (N >> 5);
      int blocks = (tiles + 7) >> 3;
      if (actout == 0)
        gemm_t<2, 2, 0, 0><<<blocks, 256, 0, stream>>>(A, B, C_, M, N, K, lda, ldb, ldc);
      else if (actout == 1)
        gemm_t<2, 2, 0, 1><<<blocks, 256, 0, stream>>>(A, B, C_, M, N, K, lda, ldb, ldc);
      else
        gemm_t<2, 2, 0, 2><<<blocks, 256, 0, stream>>>(A, B, C_, M, N, K, lda, ldb, ldc);
    } else {
      int tiles = (M >> 4) * (N >> 4);
      int blocks = (tiles + 7) >> 3;
      gemm_t<1, 1, 1, 0><<<blocks, 256, 0, stream>>>(A, B, C_, M, N, K, lda, ldb, ldc);
    }
  };

  // c = ref_feats @ w_feat + (LN(s) @ w_s)[tok]
  ln_affine_k<<<64, 256, 0, stream>>>(s_in, g_s, b_s, s_ln, N_TOKN, CS);
  gemm(s_ln, w_s, sproj, N_TOKN, CATOM, CS, CS, CATOM, CATOM, 0, 0);
  gemm(ref_feats, w_feat, c, N_ATOM, CATOM, CREF, CREF, CATOM, CATOM, 0, 0);
  add_sproj_k<<<(N_ATOM * CATOM) / 256, 256, 0, stream>>>(c, sproj, tok_idx);

  // z_p = LN(z) @ w_z;  cq/ck terms;  fused pair features + MLP
  zproj_k<<<(N_TOKN * N_TOKN + 7) / 8, 256, 0, stream>>>(z_in, g_z, b_z, w_z, zp,
                                                         N_TOKN * N_TOKN);
  gemm(c, w_cq, cqt, N_ATOM, CPAIR, CATOM, CATOM, CPAIR, CPAIR, 1, 0);
  gemm(c, w_ck, ckt, N_ATOM, CPAIR, CATOM, CATOM, CPAIR, CPAIR, 1, 0);
  pair_k<<<(N_ATOM * NKW) / 256, 256, 0, stream>>>(ref_pos, tok_idx, zp, cqt, ckt,
                                                   w_dist, w_invd, w_mask,
                                                   w_p1, w_p2, w_p3, p);
  // q = c + r_l @ w_r
  q_init_k<<<(N_ATOM * CATOM) / 256, 256, 0, stream>>>(c, r_l, w_r, q);

  const int nrow_blk = (N_ATOM + 7) / 8;
  for (int l = 0; l < 3; ++l) {
    const size_t o128 = (size_t)l * CATOM * CATOM;
    const size_t o256 = (size_t)l * CATOM * 2 * CATOM;
    // attention conditioning
    gemm(c, ada_att_g + o128, agb,   N_ATOM, CATOM, CATOM, CATOM, CATOM, CATOM, 0, 2);
    gemm(c, ada_att_b + o128, abb,   N_ATOM, CATOM, CATOM, CATOM, CATOM, CATOM, 0, 0);
    gemm(c, gate_att  + o128, gateb, N_ATOM, CATOM, CATOM, CATOM, CATOM, CATOM, 0, 2);
    adaln_k<<<nrow_blk, 256, 0, stream>>>(q, agb, abb, abuf, N_ATOM);
    // projections
    gemm(abuf, wq + o128, qhb, N_ATOM, CATOM, CATOM, CATOM, CATOM, CATOM, 0, 0);
    gemm(abuf, wk + o128, khb, N_ATOM, CATOM, CATOM, CATOM, CATOM, CATOM, 0, 0);
    gemm(abuf, wv + o128, vhb, N_ATOM, CATOM, CATOM, CATOM, CATOM, CATOM, 0, 0);
    gemm(abuf, wg + o128, gsb, N_ATOM, CATOM, CATOM, CATOM, CATOM, CATOM, 0, 2);
    // local attention
    attn_k<<<N_ATOM / NQW, 128, NHEAD * NQW * NKW * sizeof(float), stream>>>(
        qhb, khb, vhb, gsb, p, wbp + (size_t)l * CPAIR * NHEAD, otb);
    gemm(otb, wo + o128, tmpb, N_ATOM, CATOM, CATOM, CATOM, CATOM, CATOM, 0, 0);
    axpy_gate_k<<<(N_ATOM * CATOM) / 256, 256, 0, stream>>>(q, tmpb, gateb,
                                                            N_ATOM * CATOM);
    // conditioned transition (SwiGLU)
    gemm(c, ada_tr_g + o128, agb,   N_ATOM, CATOM, CATOM, CATOM, CATOM, CATOM, 0, 2);
    gemm(c, ada_tr_b + o128, abb,   N_ATOM, CATOM, CATOM, CATOM, CATOM, CATOM, 0, 0);
    gemm(c, gate_tr  + o128, gateb, N_ATOM, CATOM, CATOM, CATOM, CATOM, CATOM, 0, 2);
    adaln_k<<<nrow_blk, 256, 0, stream>>>(q, agb, abb, abuf, N_ATOM);
    gemm(abuf, w1 + o256, h1b, N_ATOM, 2 * CATOM, CATOM, CATOM, 2 * CATOM, 2 * CATOM, 0, 0);
    gemm(abuf, w2 + o256, h2b, N_ATOM, 2 * CATOM, CATOM, CATOM, 2 * CATOM, 2 * CATOM, 0, 0);
    silu_mul_k<<<(N_ATOM * 2 * CATOM) / 256, 256, 0, stream>>>(h1b, h2b,
                                                               N_ATOM * 2 * CATOM);
    gemm(h1b, w3 + o256, tmpb, N_ATOM, CATOM, 2 * CATOM, 2 * CATOM, CATOM, CATOM, 0, 0);
    axpy_gate_k<<<(N_ATOM * CATOM) / 256, 256, 0, stream>>>(q, tmpb, gateb,
                                                            N_ATOM * CATOM);
  }

  // token aggregation
  gemm(q, w_qout, tbuf, N_ATOM, CTOK, CATOM, CATOM, CTOK, CTOK, 0, 1);
  segmean_k<<<N_TOKN, 128, 0, stream>>>(tbuf, tok_idx, atok);
}
